// Self_Attn_P_10685878633281
// MI455X (gfx1250) — matrix operand order repaired and checked by reading, not compiled, not run on hardware
//
#include <hip/hip_runtime.h>

// ---------------------------------------------------------------------------
// Types for CDNA5 WMMA (wave32): v_wmma_f32_16x16x32_bf16
// ---------------------------------------------------------------------------
typedef __attribute__((ext_vector_type(16))) __bf16 v16bf;
typedef __attribute__((ext_vector_type(8)))  float  v8f;
typedef __attribute__((ext_vector_type(4)))  unsigned int v4u;
typedef __attribute__((ext_vector_type(8)))  int v8i;
typedef __attribute__((ext_vector_type(4)))  int v4i;

union FragU {
    uint4 q[2];
    v16bf v;
};

static __device__ __forceinline__ v8f wmma_bf16(v16bf a, v16bf b, v8f c) {
    // 8 args: (neg_a, A, neg_b, B, c_mod, C, reuse_a, reuse_b)
    return __builtin_amdgcn_wmma_f32_16x16x32_bf16(
        false, a, false, b, (short)0, c, false, false);
}

static __device__ __forceinline__ unsigned short f2bf(float f) {
    unsigned u = __builtin_bit_cast(unsigned, f);
    u += 0x7FFFu + ((u >> 16) & 1u);   // round-to-nearest-even
    return (unsigned short)(u >> 16);
}

#define CDIM 256
#define BATCH 8
#define NQ 4096   // 64*64
#define MKV 1024  // 32*32

#define KROW 264  // K-tile LDS row stride in halfs (32 rows): 528B, 16B aligned
#define VROW 40   // V-tile LDS row stride in halfs (256 rows): 80B, 16B aligned

// ---------------------------------------------------------------------------
// Pack activation f32 [B, C, 4096] -> bf16 [B, 4096, C]   (shift/mask only)
// ---------------------------------------------------------------------------
__global__ __launch_bounds__(256)
void pack_act(const float* __restrict__ in, unsigned short* __restrict__ outb) {
    size_t idx = (size_t)blockIdx.x * 256 + threadIdx.x;
    int c = (int)(idx & 255);
    size_t t = idx >> 8;
    int p = (int)(t & (NQ - 1));
    int b = (int)(t >> 12);
    outb[idx] = f2bf(in[((size_t)b * CDIM + c) * NQ + p]);
}

// ---------------------------------------------------------------------------
// Pack weights f32 [O=256, I=256, T] -> bf16 [T, O, I]
// ---------------------------------------------------------------------------
__global__ __launch_bounds__(256)
void pack_w(const float* __restrict__ w, unsigned short* __restrict__ outb, int T) {
    size_t idx = (size_t)blockIdx.x * 256 + threadIdx.x;
    size_t total = (size_t)T * CDIM * CDIM;
    if (idx >= total) return;
    int i = (int)(idx & 255);
    size_t t2 = idx >> 8;
    int o = (int)(t2 & 255);
    int t = (int)(t2 >> 8);
    outb[idx] = f2bf(w[((size_t)o * CDIM + i) * T + t]);
}

// ---------------------------------------------------------------------------
// Conv as tap-decomposed implicit GEMM using WMMA bf16. Input fixed 64x64.
//   srcT : bf16 [B, 4096, 256]      (A side, rows = output positions)
//   wt   : bf16 [taps, 256, 256]    (B side, lane = out channel, K = in ch)
//   outp : f32  [B, outN, 256]      (pre-GN, bias added)
// WG = 8 waves arranged 4(m) x 2(n): tile 128 rows x 64 cols.
// Tap loop is nested kh/kw (no runtime div/mod anywhere in the hot loop).
// ---------------------------------------------------------------------------
__global__ __launch_bounds__(256)
void conv_gemm_wmma(const unsigned short* __restrict__ srcT,
                    const unsigned short* __restrict__ wt,
                    const float* __restrict__ biasp,
                    float* __restrict__ outp,
                    int outWsh, int kdim, int stride, int pad) {
    const int lane = threadIdx.x & 31;
    const int wave = threadIdx.x >> 5;
    const int wm = wave >> 1;          // 0..3
    const int wn = wave & 1;           // 0..1
    const int half = lane >> 4;
    const int l16 = lane & 15;

    const int outW = 1 << outWsh;
    const int outN = outW * outW;
    const int rowBase = blockIdx.x * 128 + wm * 32;
    const int colBase = blockIdx.y * 64 + wn * 32;
    const int b = blockIdx.z;

    // Per-subtile output coordinates (shift/mask; no division).
    int hB[2], wB[2];
#pragma unroll
    for (int i = 0; i < 2; ++i) {
        int rowi = rowBase + i * 16 + l16;
        hB[i] = (rowi >> outWsh) * stride;
        wB[i] = (rowi & (outW - 1)) * stride;
    }
    const unsigned short* srcB = srcT + (size_t)b * NQ * CDIM;

    v8f acc[2][2] = {};

    const unsigned short* wtap = wt;   // advances by 256*256 per tap
    for (int kh = 0; kh < kdim; ++kh) {
        const int dh = kh - pad;
        for (int kw = 0; kw < kdim; ++kw, wtap += CDIM * CDIM) {
            const int dw = kw - pad;

            const unsigned short* aptr[2];
            bool avalid[2];
#pragma unroll
            for (int i = 0; i < 2; ++i) {
                int hs = hB[i] + dh, ws = wB[i] + dw;
                avalid[i] = ((unsigned)hs < 64u) && ((unsigned)ws < 64u);
                aptr[i] = srcB + (size_t)((hs << 6) + ws) * CDIM;
            }
            const unsigned short* bptr[2];
#pragma unroll
            for (int j = 0; j < 2; ++j)
                bptr[j] = wtap + (size_t)(colBase + j * 16 + l16) * CDIM;

#pragma unroll
            for (int kc = 0; kc < 8; ++kc) {
                FragU a[2], bb[2];
#pragma unroll
                for (int i = 0; i < 2; ++i) {
                    if (avalid[i]) {
                        // A 16x32 bf16 layout: K halves at half*8, 16+half*8
                        a[i].q[0] = *(const uint4*)(aptr[i] + kc * 32 + half * 8);
                        a[i].q[1] = *(const uint4*)(aptr[i] + kc * 32 + 16 + half * 8);
                    } else {
                        a[i].q[0] = uint4{0u, 0u, 0u, 0u};
                        a[i].q[1] = uint4{0u, 0u, 0u, 0u};
                    }
                }
#pragma unroll
                for (int j = 0; j < 2; ++j) {
                    // B 32x16 bf16 layout: K halves at half*8, 16+half*8
                    bb[j].q[0] = *(const uint4*)(bptr[j] + kc * 32 + half * 8);
                    bb[j].q[1] = *(const uint4*)(bptr[j] + kc * 32 + 16 + half * 8);
                }
#pragma unroll
                for (int i = 0; i < 2; ++i)
#pragma unroll
                    for (int j = 0; j < 2; ++j)
                        acc[i][j] = wmma_bf16(a[i].v, bb[j].v, acc[i][j]);
            }
        }
    }

    // Epilogue: add bias, store f32. D layout: row = r + 8*half, col = l16.
#pragma unroll
    for (int i = 0; i < 2; ++i)
#pragma unroll
        for (int j = 0; j < 2; ++j) {
            int col = colBase + j * 16 + l16;
            float bv = biasp[col];
#pragma unroll
            for (int r = 0; r < 8; ++r) {
                int rowi = rowBase + i * 16 + r + 8 * half;
                outp[((size_t)b * outN + rowi) * CDIM + col] = acc[i][j][r] + bv;
            }
        }
}

// ---------------------------------------------------------------------------
// GroupNorm(32) statistics: one WG per (b, group); group = 8 channels.
// ---------------------------------------------------------------------------
__global__ __launch_bounds__(256)
void gn_stats(const float* __restrict__ pre, float* __restrict__ stats, int NN) {
    int bg = blockIdx.x;
    int g = bg & 31, b = bg >> 5;
    const float* base = pre + (size_t)b * NN * CDIM + g * 8;
    float s = 0.f, s2 = 0.f;
    for (int n = threadIdx.x; n < NN; n += 256) {
        const float* p = base + (size_t)n * CDIM;
#pragma unroll
        for (int j = 0; j < 8; ++j) { float v = p[j]; s += v; s2 += v * v; }
    }
    __shared__ float sh0[256], sh1[256];
    sh0[threadIdx.x] = s; sh1[threadIdx.x] = s2;
    __syncthreads();
    for (int off = 128; off > 0; off >>= 1) {
        if ((int)threadIdx.x < off) {
            sh0[threadIdx.x] += sh0[threadIdx.x + off];
            sh1[threadIdx.x] += sh1[threadIdx.x + off];
        }
        __syncthreads();
    }
    if (threadIdx.x == 0) {
        float cnt = (float)NN * 8.0f;
        float mu = sh0[0] / cnt;
        float var = sh1[0] / cnt - mu * mu;
        stats[bg * 2] = mu;
        stats[bg * 2 + 1] = rsqrtf(var + 1e-5f);
    }
}

// ---------------------------------------------------------------------------
// GroupNorm apply + SiLU + bf16 cast. NN = 1<<nsh (shift/mask only).
// mode 0: out bf16 [B, NN, 256];  mode 1: out bf16 [B, 256, NN]
// ---------------------------------------------------------------------------
__global__ __launch_bounds__(256)
void gn_apply(const float* __restrict__ pre, const float* __restrict__ stats,
              const float* __restrict__ scale, const float* __restrict__ bias,
              unsigned short* __restrict__ outb, int nsh, int transposeOut) {
    const int NN = 1 << nsh;
    size_t idx = (size_t)blockIdx.x * 256 + threadIdx.x;
    int c = (int)(idx & 255);
    size_t nb = idx >> 8;
    int n = (int)(nb & (size_t)(NN - 1));
    int b = (int)(nb >> nsh);
    int g = c >> 3;
    float mu = stats[(b * 32 + g) * 2];
    float rs = stats[(b * 32 + g) * 2 + 1];
    float yv = (pre[idx] - mu) * rs * scale[c] + bias[c];
    float sv = yv / (1.f + __expf(-yv));          // SiLU
    unsigned short h = f2bf(sv);
    if (transposeOut) outb[((size_t)b * CDIM + c) * NN + n] = h;
    else outb[idx] = h;
}

// ---------------------------------------------------------------------------
// TDM: load V tile [256 rows x 32 cols] bf16 from vn[B,256,1024] into LDS,
// with TDM-applied padding (64B data + 16B pad => 80B LDS rows).
// One instruction per WG (issued by wave 0); tracked with TENSORcnt.
// clang-23 toolchain: 6-arg builtin (g0 v4u, g1 v8i, g2 v4i, g3 v4i, v8i, cpol).
// ---------------------------------------------------------------------------
static __device__ __forceinline__ void tdm_load_v(unsigned ldsBase,
                                                  const unsigned short* gptr) {
    unsigned long long ga = (unsigned long long)(uintptr_t)gptr;
    v4u g0;
    g0.x = 1u;                                   // count=1 (valid descriptor)
    g0.y = ldsBase;                              // lds_addr (bytes)
    g0.z = (unsigned)ga;                         // global_addr[31:0]
    g0.w = (unsigned)(ga >> 32) | 0x80000000u;   // global_addr[56:32] | type=2
    v8i g1;
    g1[0] = (int)((1u << 16)      // data_size = 1 -> 2 bytes
                | (1u << 20)      // pad_enable
                | (3u << 22)      // pad_interval: 16 DWORDs = 64B (one row)
                | (3u << 25));    // pad_amount: 4 DWORDs = 16B
    g1[1] = (int)(1024u << 16);   // tensor_dim0 = 1024 (lo16 at bits [63:48])
    g1[2] = (int)(256u << 16);    // tensor_dim0 hi=0, tensor_dim1 = 256
    g1[3] = (int)(32u << 16);     // tensor_dim1 hi=0, tile_dim0 = 32
    g1[4] = 256;                  // tile_dim1 = 256
    g1[5] = 1024;                 // tensor_dim0_stride = 1024
    g1[6] = 0;
    g1[7] = 0;
    v4i gz4 = {0, 0, 0, 0};
    v8i gz8 = {0, 0, 0, 0, 0, 0, 0, 0};
    __builtin_amdgcn_tensor_load_to_lds(g0, g1, gz4, gz4, gz8, 0);
}

// ---------------------------------------------------------------------------
// Flash attention: per wave 16 query rows, K/V swept in 32-wide tiles.
// K tile double-buffered in LDS via global_load_async_to_lds_b128 (ASYNCcnt),
// V tile double-buffered in LDS via TENSOR_LOAD_TO_LDS (TENSORcnt).
//   qn: bf16 [B, 4096, 256]   kn: bf16 [B, 1024, 256]   vn: bf16 [B, 256, 1024]
//   out: f32 [B, 256, 4096]  (NCHW), scaled by gamma.
// Block = 128 threads = 4 waves; grid = (4096/64, B).
// ---------------------------------------------------------------------------
__global__ __launch_bounds__(128)
void attn_flash(const unsigned short* __restrict__ qn,
                const unsigned short* __restrict__ kn,
                const unsigned short* __restrict__ vn,
                const float* __restrict__ gamma,
                float* __restrict__ out) {
    __shared__ __align__(16) unsigned short kTile[2][32 * KROW];   // 33,792 B
    __shared__ __align__(16) unsigned short vTile[2][256 * VROW];  // 40,960 B
    __shared__ __align__(16) unsigned short pTile[4][16 * VROW];   //  5,120 B

    const int lane = threadIdx.x & 31;
    const int wave = threadIdx.x >> 5;
    const int half = lane >> 4;
    const int l16 = lane & 15;
    const int b = blockIdx.y;
    const int nBase = blockIdx.x * 64 + wave * 16;
    const int row = nBase + l16;

    const unsigned short* kb = kn + (size_t)b * MKV * CDIM;
    const unsigned short* vb = vn + (size_t)b * CDIM * MKV;

    // Per-lane K-tile async-copy assignment: 4 waves x 8 rows, 4 lanes/row.
    const int rowK = wave * 8 + (lane >> 2);
    const int cb = lane & 3;
    auto issueK = [&](int it, int buf) {
        const unsigned short* src = kb + (size_t)(it * 32 + rowK) * CDIM + cb * 8;
        unsigned dst = (unsigned)(uintptr_t)&kTile[buf][rowK * KROW + cb * 8];
#pragma unroll
        for (int i = 0; i < 8; ++i) {
            asm volatile("global_load_async_to_lds_b128 %0, %1, off"
                         :: "v"(dst + (unsigned)(i * 64)),
                            "v"((unsigned long long)(uintptr_t)(src + i * 32))
                         : "memory");
        }
    };
    auto issueV = [&](int it, int buf) {
        if (wave == 0)
            tdm_load_v((unsigned)(uintptr_t)&vTile[buf][0], vb + it * 32);
    };

    // Resident Q fragments for K=256 (8 k-steps).
    v16bf qa[8];
    const unsigned short* qrow = qn + ((size_t)b * NQ + row) * CDIM;
#pragma unroll
    for (int kc = 0; kc < 8; ++kc) {
        FragU u;
        u.q[0] = *(const uint4*)(qrow + kc * 32 + half * 8);
        u.q[1] = *(const uint4*)(qrow + kc * 32 + 16 + half * 8);
        qa[kc] = u.v;
    }

    v8f O[16] = {};
    float rmax[8], rsum[8];
#pragma unroll
    for (int r = 0; r < 8; ++r) { rmax[r] = -__builtin_inff(); rsum[r] = 0.f; }

    unsigned short* myP = &pTile[wave][0];

    issueK(0, 0);
    issueV(0, 0);

    for (int it = 0; it < 32; ++it) {
        // Own async/tensor transfers for buf[it&1] complete, then publish.
        asm volatile("s_wait_asynccnt 0x0" ::: "memory");
        if (wave == 0) __builtin_amdgcn_s_wait_tensorcnt(0);
        __syncthreads();
        // Prefetch next tiles; the barrier above proves every wave finished
        // reading buf[(it+1)&1] (last used at iteration it-1).
        if (it + 1 < 32) {
            issueK(it + 1, (it + 1) & 1);
            issueV(it + 1, (it + 1) & 1);
        }

        const unsigned short* kt = &kTile[it & 1][0];
        const unsigned short* vt = &vTile[it & 1][0];

        v8f S0 = {}, S1 = {};
#pragma unroll
        for (int kc = 0; kc < 8; ++kc) {
            FragU u0, u1;
            u0.q[0] = *(const uint4*)(kt + l16 * KROW + kc * 32 + half * 8);
            u0.q[1] = *(const uint4*)(kt + l16 * KROW + kc * 32 + 16 + half * 8);
            u1.q[0] = *(const uint4*)(kt + (16 + l16) * KROW + kc * 32 + half * 8);
            u1.q[1] = *(const uint4*)(kt + (16 + l16) * KROW + kc * 32 + 16 + half * 8);
            S0 = wmma_bf16(qa[kc], u0.v, S0);
            S1 = wmma_bf16(qa[kc], u1.v, S1);
        }

        // Online softmax. D layout: lane holds col=l16, rows r+8*half.
        float alpha[8];
#pragma unroll
        for (int r = 0; r < 8; ++r) {
            float v0 = S0[r] * 0.0625f;     // 1/sqrt(256)
            float v1 = S1[r] * 0.0625f;
            float mx = fmaxf(v0, v1);
            mx = fmaxf(mx, __shfl_xor(mx, 1, 32));
            mx = fmaxf(mx, __shfl_xor(mx, 2, 32));
            mx = fmaxf(mx, __shfl_xor(mx, 4, 32));
            mx = fmaxf(mx, __shfl_xor(mx, 8, 32));
            float nm = fmaxf(rmax[r], mx);
            alpha[r] = __expf(rmax[r] - nm);
            rmax[r] = nm;
            float p0 = __expf(v0 - nm);
            float p1 = __expf(v1 - nm);
            S0[r] = p0; S1[r] = p1;
            float rsl = p0 + p1;
            rsl += __shfl_xor(rsl, 1, 32);
            rsl += __shfl_xor(rsl, 2, 32);
            rsl += __shfl_xor(rsl, 4, 32);
            rsl += __shfl_xor(rsl, 8, 32);
            rsum[r] = rsum[r] * alpha[r] + rsl;
        }
#pragma unroll
        for (int t = 0; t < 16; ++t)
#pragma unroll
            for (int r = 0; r < 8; ++r)
                O[t][r] *= alpha[r];

        // P (D layout) -> LDS -> A-fragment layout (wave-private region; DS is
        // in-order per wave, explicit s_wait_dscnt before the reads).
#pragma unroll
        for (int r = 0; r < 8; ++r) {
            myP[(r + 8 * half) * VROW + l16]      = f2bf(S0[r]);
            myP[(r + 8 * half) * VROW + 16 + l16] = f2bf(S1[r]);
        }
        asm volatile("s_wait_dscnt 0x0" ::: "memory");
        FragU pu;
        pu.q[0] = *(const uint4*)(myP + l16 * VROW + half * 8);
        pu.q[1] = *(const uint4*)(myP + l16 * VROW + 16 + half * 8);
        v16bf pa = pu.v;

        // O += P x V  (16 column tiles of 16 channels), V from LDS.
#pragma unroll
        for (int ct = 0; ct < 16; ++ct) {
            FragU u;
            const unsigned short* vrow = vt + (ct * 16 + l16) * VROW;
            u.q[0] = *(const uint4*)(vrow + half * 8);
            u.q[1] = *(const uint4*)(vrow + 16 + half * 8);
            O[ct] = wmma_bf16(pa, u.v, O[ct]);
        }
    }

    const float g = gamma[0];
    float inv[8];
#pragma unroll
    for (int r = 0; r < 8; ++r) inv[r] = g / rsum[r];
#pragma unroll
    for (int ct = 0; ct < 16; ++ct) {
        int c = ct * 16 + l16;
        float* op = out + ((size_t)b * CDIM + c) * NQ;
#pragma unroll
        for (int r = 0; r < 8; ++r) {
            int n = nBase + r + 8 * half;
            op[n] = O[ct][r] * inv[r];
        }
    }
}

// ---------------------------------------------------------------------------
// Host orchestration
// ---------------------------------------------------------------------------
extern "C" void kernel_launch(void* const* d_in, const int* in_sizes, int n_in,
                              void* d_out, int out_size, void* d_ws, size_t ws_size,
                              hipStream_t stream) {
    const float* x        = (const float*)d_in[0];
    const float* y        = (const float*)d_in[1];
    const float* Wq       = (const float*)d_in[2];
    const float* bq       = (const float*)d_in[3];
    const float* gq_scale = (const float*)d_in[4];
    const float* gq_bias  = (const float*)d_in[5];
    const float* Wk       = (const float*)d_in[6];
    const float* bk       = (const float*)d_in[7];
    const float* gk_scale = (const float*)d_in[8];
    const float* gk_bias  = (const float*)d_in[9];
    const float* Wv       = (const float*)d_in[10];
    const float* bv       = (const float*)d_in[11];
    const float* gv_scale = (const float*)d_in[12];
    const float* gv_bias  = (const float*)d_in[13];
    const float* gamma    = (const float*)d_in[14];
    float* outp = (float*)d_out;

    char* ws = (char*)d_ws;
    size_t off = 0;
    auto alloc = [&](size_t bytes) {
        char* p = ws + off;
        off = (off + bytes + 255) & ~(size_t)255;
        return p;
    };
    unsigned short* xT    = (unsigned short*)alloc((size_t)BATCH * NQ  * CDIM * 2);
    unsigned short* yT    = (unsigned short*)alloc((size_t)BATCH * NQ  * CDIM * 2);
    unsigned short* wqt   = (unsigned short*)alloc((size_t)9 * CDIM * CDIM * 2);
    unsigned short* wkt   = (unsigned short*)alloc((size_t)4 * CDIM * CDIM * 2);
    unsigned short* wvt   = (unsigned short*)alloc((size_t)4 * CDIM * CDIM * 2);
    float* q_pre          = (float*)alloc((size_t)BATCH * NQ  * CDIM * 4);
    float* k_pre          = (float*)alloc((size_t)BATCH * MKV * CDIM * 4);
    float* v_pre          = (float*)alloc((size_t)BATCH * MKV * CDIM * 4);
    float* st_q           = (float*)alloc((size_t)BATCH * 32 * 2 * 4);
    float* st_k           = (float*)alloc((size_t)BATCH * 32 * 2 * 4);
    float* st_v           = (float*)alloc((size_t)BATCH * 32 * 2 * 4);
    // Reuse packed-activation buffers (dead after convs) for normalized tensors.
    unsigned short* qn = xT;                                    // [B,4096,256]
    unsigned short* kn = yT;                                    // [B,1024,256]
    unsigned short* vn = yT + (size_t)BATCH * MKV * CDIM;       // [B,256,1024]
    (void)ws_size; (void)n_in; (void)in_sizes; (void)out_size;

    // 1) Pack activations & weights to bf16 / GEMM-friendly layouts.
    pack_act<<<32768, 256, 0, stream>>>(x, xT);
    pack_act<<<32768, 256, 0, stream>>>(y, yT);
    pack_w<<<2304, 256, 0, stream>>>(Wq, wqt, 9);
    pack_w<<<1024, 256, 0, stream>>>(Wk, wkt, 4);
    pack_w<<<1024, 256, 0, stream>>>(Wv, wvt, 4);

    // 2) Convs as WMMA implicit GEMMs (bias folded in).
    conv_gemm_wmma<<<dim3(32, 4, BATCH), 256, 0, stream>>>(
        xT, wqt, bq, q_pre, 6, 3, 1, 1);
    conv_gemm_wmma<<<dim3(8, 4, BATCH), 256, 0, stream>>>(
        yT, wkt, bk, k_pre, 5, 2, 2, 0);
    conv_gemm_wmma<<<dim3(8, 4, BATCH), 256, 0, stream>>>(
        yT, wvt, bv, v_pre, 5, 2, 2, 0);

    // 3) GroupNorm stats + apply(SiLU, bf16).
    gn_stats<<<BATCH * 32, 256, 0, stream>>>(q_pre, st_q, NQ);
    gn_stats<<<BATCH * 32, 256, 0, stream>>>(k_pre, st_k, MKV);
    gn_stats<<<BATCH * 32, 256, 0, stream>>>(v_pre, st_v, MKV);
    gn_apply<<<32768, 256, 0, stream>>>(q_pre, st_q, gq_scale, gq_bias, qn, 12, 0);
    gn_apply<<<8192,  256, 0, stream>>>(k_pre, st_k, gk_scale, gk_bias, kn, 10, 0);
    gn_apply<<<8192,  256, 0, stream>>>(v_pre, st_v, gv_scale, gv_bias, vn, 10, 1);

    // 4) Flash attention (QK^T softmax, PV) entirely on WMMA,
    //    async-LDS K staging + TDM V staging, double-buffered.
    attn_flash<<<dim3(64, BATCH), 128, 0, stream>>>(qn, kn, vn, gamma, outp);
}
